// GCN_3650722202369
// MI455X (gfx1250) — matrix operand order repaired and checked by reading, not compiled, not run on hardware
//
#include <hip/hip_runtime.h>

typedef __attribute__((ext_vector_type(16))) __bf16 v16bf;
typedef __attribute__((ext_vector_type(8)))  float  v8f;

// ---------------------------------------------------------------------------
// Helpers
// ---------------------------------------------------------------------------
__device__ __forceinline__ unsigned pack_bf16x2(float x, float y) {
    union { __bf16 b; unsigned short s; } lo, hi;
    lo.b = (__bf16)x; hi.b = (__bf16)y;
    return ((unsigned)hi.s << 16) | (unsigned)lo.s;
}

// ---------------------------------------------------------------------------
// Small elementwise kernels
// ---------------------------------------------------------------------------
__global__ void k_fill1(float* p, int n) {
    int i = blockIdx.x * blockDim.x + threadIdx.x;
    if (i < n) p[i] = 1.0f;
}

__global__ void k_zero(float* p, long long n) {
    long long i = (long long)blockIdx.x * blockDim.x + threadIdx.x;
    if (i < n) p[i] = 0.0f;
}

__global__ void k_deg_accum(const int* __restrict__ dst, const float* __restrict__ ew,
                            float* __restrict__ deg, int E) {
    int e = blockIdx.x * blockDim.x + threadIdx.x;
    if (e < E) atomicAdd(&deg[dst[e]], ew[e]);
}

__global__ void k_rsqrt_inplace(float* p, int n) {
    int i = blockIdx.x * blockDim.x + threadIdx.x;
    if (i < n) {
        float d = p[i];
        p[i] = d > 0.0f ? __frsqrt_rn(d) : 0.0f;
    }
}

__global__ void k_edge_norm(const int* __restrict__ src, const int* __restrict__ dst,
                            const float* __restrict__ ew, const float* __restrict__ dinv,
                            float* __restrict__ nrm, int E) {
    int e = blockIdx.x * blockDim.x + threadIdx.x;
    if (e < E) nrm[e] = dinv[src[e]] * ew[e] * dinv[dst[e]];
}

// fp32 -> bf16, 4 elements/thread (b128 load, b64 store)
__global__ void k_cvt_bf16x4(const float* __restrict__ in, unsigned* __restrict__ out,
                             long long n4) {
    long long i = (long long)blockIdx.x * blockDim.x + threadIdx.x;
    if (i >= n4) return;
    float4 v = ((const float4*)in)[i];
    uint2 o;
    o.x = pack_bf16x2(v.x, v.y);
    o.y = pack_bf16x2(v.z, v.w);
    ((uint2*)out)[i] = o;
}

// ---------------------------------------------------------------------------
// Weight pre-swizzle: write W (fp32, K x N row-major) as bf16 pairs in exact
// per-lane B-fragment order (CDNA5 ISA 7.12.2, wave32):
//   record[(ks*NT + t)*32 + lane] = 8 dwords, dword r packs
//   K = ks*32 + half*8 + 2r + (r>=4 ? 8 : 0) (+1), N = t*16 + (lane&15)
// Each lane's fragment is then 32 contiguous bytes -> two ds_load_b128.
// ---------------------------------------------------------------------------
template <int N, int K>
__global__ void k_pack_w(const float* __restrict__ W, unsigned* __restrict__ out) {
    constexpr int NT = N / 16, KS = K / 32;
    int idx = blockIdx.x * blockDim.x + threadIdx.x;
    if (idx >= KS * NT * 32 * 8) return;
    int r    = idx & 7;
    int lane = (idx >> 3) & 31;
    int t    = (idx >> 8) % NT;
    int ks   = idx / (NT * 256);
    int half = lane >> 4, l16 = lane & 15;
    int krow = ks * 32 + half * 8 + 2 * r + (r >= 4 ? 8 : 0);
    int n    = t * 16 + l16;
    out[idx] = pack_bf16x2(W[krow * N + n], W[(krow + 1) * N + n]);
}

// ---------------------------------------------------------------------------
// bf16 WMMA GEMM:  C[M x N] = A[M x K] * W[K x N]   (fp32 accumulate)
// Block = 128 threads = 4 waves; block computes a 64-row panel across all N.
// Swizzled W image + 64-row A panel staged in LDS; every fragment is two
// aligned 16-byte LDS loads (no per-element shuffling).
// ---------------------------------------------------------------------------
template <int N, int K>
__global__ void k_gemm_bf16_wmma(const __bf16* __restrict__ A,
                                 const unsigned* __restrict__ Wsw,
                                 float* __restrict__ C, int M) {
    constexpr int NT = N / 16, KS = K / 32;
    __shared__ unsigned sW[KS * NT * 32 * 8];  // 32 KB (N=128) / 16 KB (N=64)
    __shared__ __bf16   sA[64 * K];            // 16 KB

    const int tid  = threadIdx.x;   // 0..127
    const int wave = tid >> 5;      // 0..3
    const int lane = tid & 31;
    const int half = lane >> 4;     // 0/1 (lane group)
    const int l16  = lane & 15;
    const int blockRow = blockIdx.x * 64;

    // Stage swizzled W (bulk b128 copy)
    {
        const uint4* s = (const uint4*)Wsw;
        uint4*       d = (uint4*)sW;
        constexpr int n4 = (KS * NT * 32 * 8) / 4;
        for (int i = tid; i < n4; i += 128) d[i] = s[i];
    }
    // Stage 64-row A panel (zero-fill past M)
    {
        constexpr int rowChunks = (K * 2) / 16;  // uint4 per row
        for (int i = tid; i < 64 * rowChunks; i += 128) {
            int r = i / rowChunks, cc = i % rowChunks;
            int grow = blockRow + r;
            uint4 v = make_uint4(0u, 0u, 0u, 0u);
            if (grow < M) v = ((const uint4*)(A + (size_t)grow * K))[cc];
            ((uint4*)(sA + r * K))[cc] = v;
        }
    }
    __syncthreads();

    const int mBase = wave * 16;

    union Frag { uint4 q[2]; v16bf v; };

    v8f acc[NT];
    const v8f vzero = {0.f, 0.f, 0.f, 0.f, 0.f, 0.f, 0.f, 0.f};
#pragma unroll
    for (int t = 0; t < NT; ++t) acc[t] = vzero;

#pragma unroll
    for (int ks = 0; ks < KS; ++ks) {
        const int kk = ks * 32;
        // A fragment: two contiguous 8-bf16 runs per lane
        //   K = kk + half*8 .. +7   and   K = kk + 16 + half*8 .. +7
        Frag a;
        {
            const uint4* ap = (const uint4*)(sA + (mBase + l16) * K + kk + half * 8);
            a.q[0] = ap[0];
            a.q[1] = ap[2];  // +32 bytes = +16 bf16
        }
#pragma unroll
        for (int t = 0; t < NT; ++t) {
            Frag b;
            const uint4* bp = (const uint4*)(sW + ((ks * NT + t) * 32 + lane) * 8);
            b.q[0] = bp[0];
            b.q[1] = bp[1];
            acc[t] = __builtin_amdgcn_wmma_f32_16x16x32_bf16(
                false, a.v, false, b.v, (short)0, acc[t], false, false);
        }
    }

    // D layout: VGPR r -> M = r + half*8, N = t*16 + l16
#pragma unroll
    for (int t = 0; t < NT; ++t) {
#pragma unroll
        for (int r = 0; r < 8; ++r) {
            int grow = blockRow + mBase + r + half * 8;
            if (grow < M) C[(size_t)grow * N + t * 16 + l16] = acc[t][r];
        }
    }
}

// ---------------------------------------------------------------------------
// Edge scatter-add: agg[dst, c..c+3] += norm[e] * h[src, c..c+3]
// One thread per (edge, 4 channels); working set is L2-resident (192 MB).
// ---------------------------------------------------------------------------
template <int C, int SHIFT2>  // SHIFT2 = log2(C/4)
__global__ void k_scatter4(const int* __restrict__ src, const int* __restrict__ dst,
                           const float* __restrict__ nrm, const float* __restrict__ h,
                           float* __restrict__ agg, int E) {
    long long idx = (long long)blockIdx.x * blockDim.x + threadIdx.x;
    if (idx >= (long long)E * (C / 4)) return;
    int e  = (int)(idx >> SHIFT2);
    int c4 = (int)(idx & (C / 4 - 1));
    int s = src[e], d = dst[e];
    float w = nrm[e];
    float4 v = ((const float4*)(h + (size_t)s * C))[c4];
    float* ap = agg + (size_t)d * C + c4 * 4;
    atomicAdd(ap + 0, w * v.x);
    atomicAdd(ap + 1, w * v.y);
    atomicAdd(ap + 2, w * v.z);
    atomicAdd(ap + 3, w * v.w);
}

// Layer-1 epilogue: + selfloop (dinv^2*h) + bias, ReLU, -> packed bf16 (4/thread)
__global__ void k_finish1(const float* __restrict__ agg, const float* __restrict__ h,
                          const float* __restrict__ dinv, const float* __restrict__ b,
                          unsigned* __restrict__ hb, int M) {
    int idx = blockIdx.x * blockDim.x + threadIdx.x;  // over M*32
    if (idx >= M * 32) return;
    int i = idx >> 5, c4 = idx & 31;
    float di2 = dinv[i] * dinv[i];
    float4 a  = ((const float4*)agg)[idx];
    float4 hh = ((const float4*)h)[idx];
    float4 bb = ((const float4*)b)[c4];
    float v0 = a.x + di2 * hh.x + bb.x; v0 = v0 > 0.f ? v0 : 0.f;
    float v1 = a.y + di2 * hh.y + bb.y; v1 = v1 > 0.f ? v1 : 0.f;
    float v2 = a.z + di2 * hh.z + bb.z; v2 = v2 > 0.f ? v2 : 0.f;
    float v3 = a.w + di2 * hh.w + bb.w; v3 = v3 > 0.f ? v3 : 0.f;
    uint2 o;
    o.x = pack_bf16x2(v0, v1);
    o.y = pack_bf16x2(v2, v3);
    ((uint2*)hb)[idx] = o;
}

// Layer-2 epilogue: + selfloop + bias -> fp32 output (4/thread)
__global__ void k_finish2(const float* __restrict__ agg, const float* __restrict__ h,
                          const float* __restrict__ dinv, const float* __restrict__ b,
                          float* __restrict__ out, int M) {
    int idx = blockIdx.x * blockDim.x + threadIdx.x;  // over M*16
    if (idx >= M * 16) return;
    int i = idx >> 4, c4 = idx & 15;
    float di2 = dinv[i] * dinv[i];
    float4 a  = ((const float4*)agg)[idx];
    float4 hh = ((const float4*)h)[idx];
    float4 bb = ((const float4*)b)[c4];
    float4 o;
    o.x = a.x + di2 * hh.x + bb.x;
    o.y = a.y + di2 * hh.y + bb.y;
    o.z = a.z + di2 * hh.z + bb.z;
    o.w = a.w + di2 * hh.w + bb.w;
    ((float4*)out)[idx] = o;
}

// ---------------------------------------------------------------------------
static inline char* ws_align(char* p) {
    return (char*)(((uintptr_t)p + 255u) & ~(uintptr_t)255u);
}

extern "C" void kernel_launch(void* const* d_in, const int* in_sizes, int n_in,
                              void* d_out, int out_size, void* d_ws, size_t ws_size,
                              hipStream_t stream) {
    const float* x   = (const float*)d_in[0];
    const int*   ei  = (const int*)d_in[1];
    const float* ew  = (const float*)d_in[2];
    const float* W1  = (const float*)d_in[3];
    const float* b1  = (const float*)d_in[4];
    const float* W2  = (const float*)d_in[5];
    const float* b2  = (const float*)d_in[6];
    float*       out = (float*)d_out;

    const int IN = 128, HID = 128, OUT = 64;
    const int M = in_sizes[0] / IN;   // 100000
    const int E = in_sizes[2];        // 1600000
    const int* srcIdx = ei;           // edge_index[0]
    const int* dstIdx = ei + E;       // edge_index[1]

    // Workspace carve-up (reused buffers)
    char* p = (char*)d_ws;
    float*    dinv = (float*)p;              p = ws_align(p + (size_t)M * 4);
    float*    nrm  = (float*)p;              p = ws_align(p + (size_t)E * 4);
    __bf16*   xb   = (__bf16*)p;             p = ws_align(p + (size_t)M * IN * 2);  // reused as hb
    unsigned* W1s  = (unsigned*)p;           p = ws_align(p + (size_t)IN * HID * 2); // swizzled bf16 pairs
    unsigned* W2s  = (unsigned*)p;           p = ws_align(p + (size_t)HID * OUT * 2);
    float*    h    = (float*)p;              p = ws_align(p + (size_t)M * HID * 4); // h1, then h2
    float*    agg  = (float*)p;              p = ws_align(p + (size_t)M * HID * 4); // agg1, then agg2
    __bf16*   hb   = xb;                     // alias: xb is dead after GEMM1

    const int TB = 256;
    auto blk = [&](long long n) { return (unsigned)((n + TB - 1) / TB); };

    // 1) degrees (self-loop weight 1 -> init 1.0), dinv = rsqrt(deg), edge norms
    k_fill1<<<blk(M), TB, 0, stream>>>(dinv, M);
    k_deg_accum<<<blk(E), TB, 0, stream>>>(dstIdx, ew, dinv, E);
    k_rsqrt_inplace<<<blk(M), TB, 0, stream>>>(dinv, M);
    k_edge_norm<<<blk(E), TB, 0, stream>>>(srcIdx, dstIdx, ew, dinv, nrm, E);

    // 2) feature conversion + weight pre-swizzle (fp32 -> bf16 fragment order)
    k_cvt_bf16x4<<<blk((long long)M * IN / 4), TB, 0, stream>>>(x, (unsigned*)xb,
                                                                (long long)M * IN / 4);
    k_pack_w<128, 128><<<blk(IN * HID / 2), TB, 0, stream>>>(W1, W1s);
    k_pack_w<64, 128><<<blk(HID * OUT / 2), TB, 0, stream>>>(W2, W2s);

    // 3) layer 1: h = x @ W1  (WMMA bf16)
    {
        unsigned g = (unsigned)((M + 63) / 64);
        k_gemm_bf16_wmma<128, 128><<<g, 128, 0, stream>>>(xb, W1s, h, M);
    }
    // 4) aggregate (L2-resident scatter-add), epilogue: +selfloop +b1, ReLU, ->bf16
    k_zero<<<blk((long long)M * HID), TB, 0, stream>>>(agg, (long long)M * HID);
    k_scatter4<128, 5><<<blk((long long)E * 32), TB, 0, stream>>>(srcIdx, dstIdx, nrm, h, agg, E);
    k_finish1<<<blk((long long)M * 32), TB, 0, stream>>>(agg, h, dinv, b1, (unsigned*)hb, M);

    // 5) layer 2: h2 = relu_h @ W2  (WMMA bf16), reuse h/agg buffers
    {
        unsigned g = (unsigned)((M + 63) / 64);
        k_gemm_bf16_wmma<64, 128><<<g, 128, 0, stream>>>(hb, W2s, h, M);
    }
    k_zero<<<blk((long long)M * OUT), TB, 0, stream>>>(agg, (long long)M * OUT);
    k_scatter4<64, 4><<<blk((long long)E * 16), TB, 0, stream>>>(srcIdx, dstIdx, nrm, h, agg, E);
    k_finish2<<<blk((long long)M * 16), TB, 0, stream>>>(agg, h, dinv, b2, out, M);
}
